// CharLevelModel_3659312136209
// MI455X (gfx1250) — matrix operand order repaired and checked by reading, not compiled, not run on hardware
//
#include <hip/hip_runtime.h>
#include <hip/hip_bf16.h>

// ---------------- problem constants ----------------
#define BB   256
#define TTOK 128
#define CC   141
#define DD   1536
#define HH   141
#define GG   564     // 4*H
#define LLAY 4
#define MROWS (BB*CC)        // 36096
#define MTILES (MROWS/16)    // 2256 (divisible)
#define NPAD 576             // G padded to 16*36
#define NT_G 36              // NPAD/16
#define KT_H 5               // H=141 -> 160/32
#define KT_D 48              // D=1536/32
#define KT_2H 9              // 2H=282 -> 288/32

typedef __attribute__((ext_vector_type(16))) __bf16 v16bf;
typedef __attribute__((ext_vector_type(8)))  float  v8f;

__device__ __forceinline__ unsigned short f2bf(float f) {
    unsigned int u = __float_as_uint(f);
    unsigned int r = u + 0x7FFFu + ((u >> 16) & 1u);   // round-to-nearest-even
    return (unsigned short)(r >> 16);
}
__device__ __forceinline__ float sigf(float x) { return 1.0f / (1.0f + __expf(-x)); }

// A-fragment K pattern (16-bit A 16x32, ISA 7.12.2):
//  lanes 0-15 row m: v0-3 -> K 0..7, v4-7 -> K 16..23 ; lanes 16-31: +8
__device__ __forceinline__ int kA(int e, int lh) { return (e & 7) + ((e >> 3) << 4) + lh * 8; }
// contiguous K order (unused; B is packed with kA like A): lanes 0-15 col n hold K 0..15, lanes 16-31 K 16..31
__device__ __forceinline__ int kB(int e, int lh) { return e + lh * 16; }

// packed-A element address for row r, col k (u16 units)
__device__ __forceinline__ size_t packAaddr(int r, int k, int ktiles) {
    int tm = r >> 4, ml = r & 15, tk = k >> 5, kk = k & 31;
    int lh = (kk >> 3) & 1;
    int e  = (kk & 7) + ((kk >> 4) << 3);
    return ((size_t)(tm * ktiles + tk) * 512) + (size_t)((ml + lh * 16) * 16 + e);
}
// same inverse mapping but within a single 16-row tile (returns u16 index in [0, ktiles*512))
__device__ __forceinline__ int tileAaddr(int m, int k) {
    int tk = k >> 5, kk = k & 31;
    int lh = (kk >> 3) & 1;
    int e  = (kk & 7) + ((kk >> 4) << 3);
    return tk * 512 + (m + lh * 16) * 16 + e;
}

// ---------------- weight packing: W (Nr x Kc row-major, used as x@W^T) -> B fragments
__global__ __launch_bounds__(256) void pack_weightB(const float* __restrict__ W,
                                                    int Nr, int Kc,
                                                    unsigned short* __restrict__ dst,
                                                    int ntiles, int ktiles) {
    size_t gid = (size_t)blockIdx.x * 256 + threadIdx.x;
    size_t tot = (size_t)ntiles * ktiles * 512;
    if (gid >= tot) return;
    int tile = (int)(gid >> 9);
    int le   = (int)(gid & 511);
    int lane = le >> 4, e = le & 15;
    int nt = tile / ktiles, kt = tile % ktiles;
    int lh = lane >> 4;
    int n  = nt * 16 + (lane & 15);
    int k  = kt * 32 + kA(e, lh);
    float v = (n < Nr && k < Kc) ? W[(size_t)n * Kc + k] : 0.0f;
    dst[gid] = f2bf(v);
}

// ---------------- token coverage: first token covering each char ----------------
__global__ __launch_bounds__(256) void tok_index(const int* __restrict__ om,
                                                 int* __restrict__ tokidx,
                                                 int* __restrict__ has) {
    int r = blockIdx.x * 256 + threadIdx.x;
    if (r >= MROWS) return;
    int b = r / CC, c = r % CC;
    int idx = 0, h = 0;
    for (int t = 0; t < TTOK; ++t) {
        int st = om[((size_t)b * TTOK + t) * 2 + 0];
        int en = om[((size_t)b * TTOK + t) * 2 + 1];
        if (!h && st <= c && c < en) { idx = t; h = 1; }
    }
    tokidx[r] = idx;
    has[r] = h;
}

// ---------------- gather token rows into packed-A bf16 ----------------
__global__ __launch_bounds__(256) void gather_packA(const float* __restrict__ feats,
                                                    const int* __restrict__ tokidx,
                                                    const int* __restrict__ has,
                                                    unsigned short* __restrict__ Apack) {
    size_t gid = (size_t)blockIdx.x * 256 + threadIdx.x;
    size_t tot = (size_t)MROWS * DD;
    if (gid >= tot) return;
    int r = (int)(gid / DD);
    int k = (int)(gid % DD);
    float v = 0.0f;
    if (has[r]) {
        int b = r / CC;
        v = feats[((size_t)b * TTOK + tokidx[r]) * DD + k];
    }
    Apack[packAaddr(r, k, KT_D)] = f2bf(v);
}

__global__ __launch_bounds__(256) void zero_u32(unsigned int* p, long n) {
    long i = (long)blockIdx.x * 256 + threadIdx.x;
    if (i < n) p[i] = 0u;
}

// ---------------- generic packed WMMA GEMM ----------------
// out[m,n] = sum_k A[m,k]*W[n,k] + bias[n]
// MODE 0: write fp32 to outF with stride ostride (pad cols get acc==0)
// MODE 1: write bf16 into packed-A layout of the NEXT gemm (ktilesNext); rows with
//         mask[r]==0 are forced to 0 (post-bias), matching the reference scatter.
// Out-of-range n-tiles are staged as ZEROS in LDS so the inner loop has NO branches
// around the WMMA ops (keeps accumulators pinned, no register shuffling).
template <int KTILES, int MODE>
__global__ __launch_bounds__(256) void gemm_wmma(const unsigned short* __restrict__ Apack,
                                                 const unsigned short* __restrict__ Bpack,
                                                 const float* __restrict__ bias,
                                                 int ntiles,
                                                 float* __restrict__ outF, int ostride, int Ncols,
                                                 unsigned short* __restrict__ outP, int ktilesNext,
                                                 const int* __restrict__ mask) {
    constexpr int KCHUNK = (KTILES % 8 == 0) ? 8 : KTILES;
    __shared__ __align__(32) unsigned short sB[4 * KCHUNK * 512];
    const int tid   = threadIdx.x;
    const int wave  = tid >> 5;
    const int lane  = tid & 31;
    const int mtile = blockIdx.x * 8 + wave;
    const int nt0   = blockIdx.y * 4;

    v8f zero = {0.f,0.f,0.f,0.f,0.f,0.f,0.f,0.f};
    v8f acc[4];
#pragma unroll
    for (int i = 0; i < 4; ++i) acc[i] = zero;

#pragma unroll 1
    for (int kt0 = 0; kt0 < KTILES; kt0 += KCHUNK) {
        __syncthreads();
        // stage 4 B-tile columns for this k-chunk into LDS; zero-fill invalid n-tiles
        constexpr int NLOAD = 4 * KCHUNK * 256;          // u32 count
        for (int q = tid; q < NLOAD; q += 256) {
            int i  = q / (KCHUNK * 256);
            int rl = q % (KCHUNK * 256);
            int nt = nt0 + i;
            unsigned int v = 0u;
            if (nt < ntiles)
                v = ((const unsigned int*)Bpack)[((size_t)nt * KTILES + kt0) * 256 + rl];
            ((unsigned int*)sB)[i * (KCHUNK * 256) + rl] = v;
        }
        __syncthreads();
#pragma unroll
        for (int ktl = 0; ktl < KCHUNK; ++ktl) {
            v16bf a = *(const v16bf*)(Apack +
                        ((size_t)(mtile * KTILES + kt0 + ktl) * 512 + lane * 16));
#pragma unroll
            for (int i = 0; i < 4; ++i) {
                v16bf b = *(const v16bf*)(sB + ((i * KCHUNK + ktl) * 512 + lane * 16));
                acc[i] = __builtin_amdgcn_wmma_f32_16x16x32_bf16(
                    false, a, false, b, (short)0, acc[i], false, false);
            }
        }
    }

    const int lh = lane >> 4, nl = lane & 15;
#pragma unroll
    for (int i = 0; i < 4; ++i) {
        int nt = nt0 + i;
        if (nt >= ntiles) continue;
#pragma unroll
        for (int j = 0; j < 8; ++j) {
            int m = mtile * 16 + j + lh * 8;
            int n = nt * 16 + nl;
            float v = acc[i][j];
            float bv = (bias && n < Ncols) ? bias[n] : 0.0f;
            v += bv;
            if (MODE == 0) {
                outF[(size_t)m * ostride + n] = v;
            } else {
                if (mask && mask[m] == 0) v = 0.0f;
                outP[packAaddr(m, n, ktilesNext)] = f2bf(v);
            }
        }
    }
}

// ---------------- bidirectional LSTM, one (layer, direction) per launch ----------------
// Each block owns 16 batch rows. The h-state lives in LDS in bf16 *A-fragment layout*
// (shA), so every wave's WMMA A-operand is two ds_load_b128 with zero conversion work.
// c-state (fp32) and gate pre-activations (fp32) also live in LDS.
// w_hh stays in global as packed B fragments (L2-resident, shared by all blocks).
__global__ __launch_bounds__(288) void lstm_dir(const float* __restrict__ xW,      // [B*C, 576]
                                                const unsigned short* __restrict__ whhP,
                                                const float* __restrict__ h0,
                                                const float* __restrict__ c0,
                                                unsigned short* __restrict__ xcatP, // packed-A, K=288
                                                float* __restrict__ hnOut,
                                                float* __restrict__ cnOut,
                                                int dir, int stateIdx) {
    __shared__ __align__(32) unsigned short shA[KT_H * 512]; // h as bf16 A-fragments (pads 0)
    __shared__ float sc[16 * 160];                           // c state (fp32, pads 0)
    __shared__ float sg[16 * NPAD];                          // gate pre-activations

    const int tid  = threadIdx.x;
    const int b0   = blockIdx.x * 16;
    const int wave = tid >> 5;
    const int lane = tid & 31;
    const int lh   = lane >> 4, nl = lane & 15;

    // init h (bf16 fragments) and c (fp32)
    for (int q = tid; q < KT_H * 512; q += 288) {
        int kt = q >> 9, le = q & 511;
        int ln = le >> 4, e = le & 15;
        int m  = ln & 15;
        int k  = kt * 32 + kA(e, ln >> 4);
        float hv = 0.f;
        if (k < HH)
            hv = h0[(size_t)stateIdx * BB * HH + (size_t)(b0 + m) * HH + k];
        shA[q] = f2bf(hv);
    }
    for (int q = tid; q < 16 * 160; q += 288) {
        int m = q / 160, i = q % 160;
        float cv = 0.f;
        if (i < HH)
            cv = c0[(size_t)stateIdx * BB * HH + (size_t)(b0 + m) * HH + i];
        sc[q] = cv;
    }
    __syncthreads();

#pragma unroll 1
    for (int s = 0; s < CC; ++s) {
        const int tt = dir ? (CC - 1 - s) : s;

        // g = xW(t) + h @ w_hh^T ; 9 waves x 4 sequential ntiles = 36 ntiles (G pad 576)
        // One accumulator live at a time -> low VGPR pressure, no vgpr-msb switching.
#pragma unroll 1
        for (int i = 0; i < 4; ++i) {
            const int nt = wave * 4 + i;
            v8f acc;
#pragma unroll
            for (int j = 0; j < 8; ++j) {
                int m = j + lh * 8;
                acc[j] = xW[((size_t)(b0 + m) * CC + tt) * NPAD + nt * 16 + nl];
            }
#pragma unroll
            for (int kt = 0; kt < KT_H; ++kt) {
                v16bf a = *(const v16bf*)(shA + (kt * 512 + lane * 16));
                v16bf b = *(const v16bf*)(whhP + ((size_t)(nt * KT_H + kt) * 512 + lane * 16));
                acc = __builtin_amdgcn_wmma_f32_16x16x32_bf16(
                    false, a, false, b, (short)0, acc, false, false);
            }
#pragma unroll
            for (int j = 0; j < 8; ++j)
                sg[(j + lh * 8) * NPAD + nt * 16 + nl] = acc[j];
        }
        __syncthreads();

        // gates + state update + emit h into next layer's packed-A input
        for (int q = tid; q < 16 * HH; q += 288) {
            int m = q / HH, i = q % HH;
            float gi = sg[m * NPAD + i];
            float gf = sg[m * NPAD + HH + i];
            float gg = sg[m * NPAD + 2 * HH + i];
            float go = sg[m * NPAD + 3 * HH + i];
            float cp = sc[m * 160 + i];
            float c2 = sigf(gf) * cp + sigf(gi) * tanhf(gg);
            float h2 = sigf(go) * tanhf(c2);
            sc[m * 160 + i] = c2;
            unsigned short hb = f2bf(h2);
            shA[tileAaddr(m, i)] = hb;                         // next-step A fragments
            int r = (b0 + m) * CC + tt;
            xcatP[packAaddr(r, dir * HH + i, KT_2H)] = hb;     // next-layer input
            if (s == CC - 1)                                   // final hidden state
                hnOut[(size_t)stateIdx * BB * HH + (size_t)(b0 + m) * HH + i] = h2;
        }
        __syncthreads();
    }

    for (int q = tid; q < 16 * HH; q += 288) {
        int m = q / HH, i = q % HH;
        cnOut[(size_t)stateIdx * BB * HH + (size_t)(b0 + m) * HH + i] = sc[m * 160 + i];
    }
}

// ---------------- head: X1/X2 = y @ w2^T + b2 ----------------
__global__ __launch_bounds__(256) void head_kernel(const float* __restrict__ y,   // [M, 288]
                                                   const float* __restrict__ w2,  // [2, 282]
                                                   const float* __restrict__ b2,
                                                   float* __restrict__ out) {
    int r = blockIdx.x * 256 + threadIdx.x;
    if (r >= MROWS) return;
    float s0 = b2[0], s1 = b2[1];
    const float* yr = y + (size_t)r * 288;
    for (int n = 0; n < 282; ++n) {
        float v = yr[n];
        s0 += v * w2[n];
        s1 += v * w2[282 + n];
    }
    out[r] = s0;
    out[MROWS + r] = s1;
}

// ---------------- host launcher ----------------
extern "C" void kernel_launch(void* const* d_in, const int* in_sizes, int n_in,
                              void* d_out, int out_size, void* d_ws, size_t ws_size,
                              hipStream_t stream) {
    const float* tok_feats = (const float*)d_in[0];
    const int*   om        = (const int*)  d_in[1];
    const float* h0        = (const float*)d_in[2];
    const float* c0        = (const float*)d_in[3];
    const float* w_lin     = (const float*)d_in[4];
    const float* b_lin     = (const float*)d_in[5];
    const float* w_ih0_f   = (const float*)d_in[6];
    const float* w_ih0_b   = (const float*)d_in[7];
    const float* w_ih_f    = (const float*)d_in[8];
    const float* w_ih_b    = (const float*)d_in[9];
    const float* w_hh_f    = (const float*)d_in[10];
    const float* w_hh_b    = (const float*)d_in[11];
    const float* b_f       = (const float*)d_in[12];
    const float* b_b       = (const float*)d_in[13];
    const float* w1        = (const float*)d_in[14];
    const float* b1        = (const float*)d_in[15];
    const float* w2        = (const float*)d_in[16];
    const float* b2        = (const float*)d_in[17];

    char* ws = (char*)d_ws;
    size_t off = 0;
    auto carve = [&](size_t bytes) -> void* {
        void* p = ws + off;
        off = (off + bytes + 255) & ~(size_t)255;
        return p;
    };

    int*            tokidx  = (int*)carve((size_t)MROWS * 4);
    int*            has     = (int*)carve((size_t)MROWS * 4);
    unsigned short* pWlin   = (unsigned short*)carve((size_t)96 * KT_D * 512 * 2);
    unsigned short* pWih0f  = (unsigned short*)carve((size_t)NT_G * KT_D * 512 * 2);
    unsigned short* pWih0b  = (unsigned short*)carve((size_t)NT_G * KT_D * 512 * 2);
    unsigned short* pWihf[3]; unsigned short* pWihb[3];
    for (int l = 0; l < 3; ++l) pWihf[l] = (unsigned short*)carve((size_t)NT_G * KT_2H * 512 * 2);
    for (int l = 0; l < 3; ++l) pWihb[l] = (unsigned short*)carve((size_t)NT_G * KT_2H * 512 * 2);
    unsigned short* pWhhf[4]; unsigned short* pWhhb[4];
    for (int l = 0; l < 4; ++l) pWhhf[l] = (unsigned short*)carve((size_t)NT_G * KT_H * 512 * 2);
    for (int l = 0; l < 4; ++l) pWhhb[l] = (unsigned short*)carve((size_t)NT_G * KT_H * 512 * 2);
    unsigned short* pW1     = (unsigned short*)carve((size_t)18 * KT_2H * 512 * 2);
    unsigned short* pAfeat  = (unsigned short*)carve((size_t)MTILES * KT_D * 512 * 2);
    unsigned short* pAx0    = (unsigned short*)carve((size_t)MTILES * KT_D * 512 * 2);
    unsigned short* pXcat   = (unsigned short*)carve((size_t)MTILES * KT_2H * 512 * 2);
    float*          xWf     = (float*)carve((size_t)MROWS * NPAD * 4);
    float*          xWb     = (float*)carve((size_t)MROWS * NPAD * 4);
    float*          yBuf    = (float*)carve((size_t)MROWS * 288 * 4);

    float* outX1 = (float*)d_out;                 // [M]   (+X2 at MROWS)
    float* hnOut = (float*)d_out + 2 * MROWS;     // [8,B,H]
    float* cnOut = hnOut + 8 * BB * HH;           // [8,B,H]

    auto packLaunch = [&](const float* W, int Nr, int Kc, unsigned short* dst, int nt, int kt) {
        size_t tot = (size_t)nt * kt * 512;
        pack_weightB<<<dim3((unsigned)((tot + 255) / 256)), 256, 0, stream>>>(W, Nr, Kc, dst, nt, kt);
    };

    // pack all weights into WMMA B-fragment layout
    packLaunch(w_lin, DD, DD, pWlin, 96, KT_D);
    packLaunch(w_ih0_f, GG, DD, pWih0f, NT_G, KT_D);
    packLaunch(w_ih0_b, GG, DD, pWih0b, NT_G, KT_D);
    for (int l = 0; l < 3; ++l) {
        packLaunch(w_ih_f + (size_t)l * GG * 2 * HH, GG, 2 * HH, pWihf[l], NT_G, KT_2H);
        packLaunch(w_ih_b + (size_t)l * GG * 2 * HH, GG, 2 * HH, pWihb[l], NT_G, KT_2H);
    }
    for (int l = 0; l < 4; ++l) {
        packLaunch(w_hh_f + (size_t)l * GG * HH, GG, HH, pWhhf[l], NT_G, KT_H);
        packLaunch(w_hh_b + (size_t)l * GG * HH, GG, HH, pWhhb[l], NT_G, KT_H);
    }
    packLaunch(w1, 2 * HH, 2 * HH, pW1, 18, KT_2H);

    // token coverage + gather into packed-A
    tok_index<<<dim3((MROWS + 255) / 256), 256, 0, stream>>>(om, tokidx, has);
    {
        size_t tot = (size_t)MROWS * DD;
        gather_packA<<<dim3((unsigned)((tot + 255) / 256)), 256, 0, stream>>>(tok_feats, tokidx, has, pAfeat);
    }
    // zero pXcat once so pad columns (k=282..287) are clean for all layers
    {
        long n32 = (long)MTILES * KT_2H * 512 / 2;
        zero_u32<<<dim3((unsigned)((n32 + 255) / 256)), 256, 0, stream>>>((unsigned int*)pXcat, n32);
    }

    // GEMM1: x0 = mask(gather(tok_feats) @ w_lin^T + b_lin) -> packed-A bf16
    gemm_wmma<KT_D, 1><<<dim3(MTILES / 8, 96 / 4), 256, 0, stream>>>(
        pAfeat, pWlin, b_lin, 96,
        nullptr, 0, DD, pAx0, KT_D, has);

    // LSTM stack
    for (int l = 0; l < LLAY; ++l) {
        if (l == 0) {
            gemm_wmma<KT_D, 0><<<dim3(MTILES / 8, NT_G / 4), 256, 0, stream>>>(
                pAx0, pWih0f, b_f, NT_G, xWf, NPAD, GG, nullptr, 0, nullptr);
            gemm_wmma<KT_D, 0><<<dim3(MTILES / 8, NT_G / 4), 256, 0, stream>>>(
                pAx0, pWih0b, b_b, NT_G, xWb, NPAD, GG, nullptr, 0, nullptr);
        } else {
            gemm_wmma<KT_2H, 0><<<dim3(MTILES / 8, NT_G / 4), 256, 0, stream>>>(
                pXcat, pWihf[l - 1], b_f + (size_t)l * GG, NT_G, xWf, NPAD, GG, nullptr, 0, nullptr);
            gemm_wmma<KT_2H, 0><<<dim3(MTILES / 8, NT_G / 4), 256, 0, stream>>>(
                pXcat, pWihb[l - 1], b_b + (size_t)l * GG, NT_G, xWb, NPAD, GG, nullptr, 0, nullptr);
        }
        lstm_dir<<<dim3(BB / 16), 288, 0, stream>>>(xWf, pWhhf[l], h0, c0, pXcat, hnOut, cnOut, 0, 2 * l);
        lstm_dir<<<dim3(BB / 16), 288, 0, stream>>>(xWb, pWhhb[l], h0, c0, pXcat, hnOut, cnOut, 1, 2 * l + 1);
    }

    // head GEMM: y = xcat @ w1^T + b1  (ntiles=18 -> 5 y-blocks, last partially zero-padded)
    gemm_wmma<KT_2H, 0><<<dim3(MTILES / 8, 5), 256, 0, stream>>>(
        pXcat, pW1, b1, 18, yBuf, 288, 2 * HH, nullptr, 0, nullptr);

    // final 2-wide projection
    head_kernel<<<dim3((MROWS + 255) / 256), 256, 0, stream>>>(yBuf, w2, b2, outX1);
}
